// CasAttention2D_28707561406621
// MI455X (gfx1250) — matrix operand order repaired and checked by reading, not compiled, not run on hardware
//
#include <hip/hip_runtime.h>

// Problem constants (from the reference)
#define BB 4
#define NB 4096
#define KK 16
#define DD 128
#define HH 16
#define TILE_N 16
#define NWAVES 4

typedef __attribute__((ext_vector_type(16))) _Float16 v16h;
typedef __attribute__((ext_vector_type(8)))  _Float16 v8h;
typedef __attribute__((ext_vector_type(4)))  _Float16 v4h;
typedef __attribute__((ext_vector_type(8)))  float    v8f;

// ---------------------------------------------------------------------------
// LDS layout (~227KB of the 320KB WGP LDS). All B-operand weights are stored
// TRANSPOSED ([n][k]) so B fragments are two contiguous ds_load_b128 per lane.
// ---------------------------------------------------------------------------
struct alignas(16) WaveBuf {
  _Float16 kb [KK * DD];    // k[b,n] tile, f16, row-major (A operand)
  _Float16 khb[KK * DD];    // kh tile, f16 (A operand for v GEMM)
  _Float16 sb [KK * DD];    // kh - qh + ph, f16 (A operand for attn MLP)
  _Float16 gb [KK * 32];    // relu(pos@P1+pb1), K padded 16->32 (cols 16.. = 0)
  _Float16 hb [KK * 32];    // relu(s@A1+ab1),  K padded 16->32 (cols 16.. = 0)
  int      mk [KK];         // mask per kv
};

struct alignas(16) Smem {
  _Float16 wkT [DD * DD];   // Wk^T  [n][k] f16
  _Float16 wvT [DD * DD];   // Wv^T  [n][k] f16
  _Float16 wqT [DD * DD];   // Wq^T  [n][k] f16
  _Float16 woT [DD * DD];   // Wo^T  [n][k] f16
  _Float16 a1T [HH * DD];   // A1^T  [16n][128k] f16
  _Float16 a2T [DD * 32];   // A2^T  [128n][32k], k 16..31 zero
  _Float16 p2T [DD * 32];   // P2^T  [128n][32k], k 16..31 zero
  float    p1 [4 * HH];
  float    pb1[HH], pb2[DD], ab1[HH], ab2[DD], bos[DD];
  _Float16 qb [TILE_N * DD];  // q tile, f16 (A operand)
  float    qhf[TILE_N * DD];  // qh tile, f32
  _Float16 xb [TILE_N * DD];  // x tile, f16 (A operand for out GEMM)
  WaveBuf  wb [NWAVES];
};

// Intra-wave LDS producer->consumer ordering (single instruction stream per
// wave; just keep DS stores from floating past dependent DS loads).
__device__ __forceinline__ void lds_wave_fence() {
  asm volatile("s_wait_dscnt 0" ::: "memory");
}

__device__ __forceinline__ v16h cat16(v8h lo, v8h hi) {
  return __builtin_shufflevector(lo, hi, 0, 1, 2, 3, 4, 5, 6, 7, 8, 9, 10, 11,
                                 12, 13, 14, 15);
}

// A fragment: f16 16x32 (MxK), row-major source, leading dim `ld` halves.
// Lane l -> M = l&15; two contiguous 16B runs at k = kb and k = 16+kb,
// kb = (l>=16) ? 8 : 0.   => 2x ds_load_b128.
__device__ __forceinline__ v16h load_frag_a(const _Float16* a, int ld, int lane) {
  const int m  = lane & 15;
  const int kb = (lane & 16) ? 8 : 0;
  const _Float16* p = a + m * ld + kb;
  return cat16(*reinterpret_cast<const v8h*>(p),
               *reinterpret_cast<const v8h*>(p + 16));
}

// B fragment from TRANSPOSED storage: element (k,n) at bT[n*ldK + k].
// Lane l -> n = l&15; two contiguous 16B runs at k = kb and k = 16+kb,
// kb = (l>=16) ? 8 : 0   => 2x ds_load_b128.
__device__ __forceinline__ v16h load_frag_bT(const _Float16* bT, int ldK,
                                             int lane) {
  const _Float16* p = bT + (lane & 15) * ldK + ((lane & 16) >> 1);
  return cat16(*reinterpret_cast<const v8h*>(p),
               *reinterpret_cast<const v8h*>(p + 16));
}

__device__ __forceinline__ v8f wmma_f16(v16h a, v16h b, v8f c) {
  return __builtin_amdgcn_wmma_f32_16x16x32_f16(false, a, false, b, (short)0, c,
                                                false, false);
}

__global__ __launch_bounds__(128, 1) void cas_attn2d_kernel(
    const float* __restrict__ q, const float* __restrict__ k,
    const float* __restrict__ pos, const int* __restrict__ mask,
    const float* __restrict__ Wq, const float* __restrict__ Wk,
    const float* __restrict__ Wv, const float* __restrict__ P1,
    const float* __restrict__ pb1, const float* __restrict__ P2,
    const float* __restrict__ pb2, const float* __restrict__ A1,
    const float* __restrict__ ab1, const float* __restrict__ A2,
    const float* __restrict__ ab2, const float* __restrict__ Wo,
    const float* __restrict__ bo, float* __restrict__ out) {
  extern __shared__ char smem_raw[];
  Smem* s = reinterpret_cast<Smem*>(smem_raw);

  const int tid  = threadIdx.x;
  const int wave = tid >> 5;
  const int lane = tid & 31;
  const int b    = blockIdx.x / (NB / TILE_N);
  const int n0   = (blockIdx.x % (NB / TILE_N)) * TILE_N;

  // ---- Phase 0: stage weights (fp32 -> f16, TRANSPOSED), biases, q tile ----
  for (int i = tid; i < DD * DD; i += 128) {
    const int r = i >> 7, c = i & 127;   // src [k=r][n=c]
    const int t = c * DD + r;            // dst [n][k]
    s->wkT[t] = (_Float16)Wk[i];
    s->wvT[t] = (_Float16)Wv[i];
    s->wqT[t] = (_Float16)Wq[i];
    s->woT[t] = (_Float16)Wo[i];
  }
  for (int i = tid; i < DD * HH; i += 128) {
    const int r = i >> 4, c = i & 15;    // src [k=r][n=c] (128x16)
    s->a1T[c * DD + r] = (_Float16)A1[i];
  }
  for (int i = tid; i < DD * 32; i += 128) {
    const int n = i >> 5, kk = i & 31;   // dst [n][k], k padded 16->32
    s->a2T[i] = (kk < HH) ? (_Float16)A2[kk * DD + n] : (_Float16)0.f;
    s->p2T[i] = (kk < HH) ? (_Float16)P2[kk * DD + n] : (_Float16)0.f;
  }
  if (tid < 4 * HH) s->p1[tid] = P1[tid];
  if (tid < HH) { s->pb1[tid] = pb1[tid]; s->ab1[tid] = ab1[tid]; }
  if (tid < DD) { s->pb2[tid] = pb2[tid]; s->ab2[tid] = ab2[tid]; s->bos[tid] = bo[tid]; }
  {
    const float4* qp4 = reinterpret_cast<const float4*>(
        q + ((size_t)b * NB + n0) * DD);
    for (int i = tid; i < TILE_N * DD / 4; i += 128) {
      const float4 f = qp4[i];
      v4h h = {(_Float16)f.x, (_Float16)f.y, (_Float16)f.z, (_Float16)f.w};
      *reinterpret_cast<v4h*>(s->qb + i * 4) = h;
    }
  }
  // zero K-padded gb/hb once (cols 16..31 stay zero forever)
  for (int i = lane; i < KK * 32; i += 32) {
    s->wb[wave].gb[i] = (_Float16)0.f;
    s->wb[wave].hb[i] = (_Float16)0.f;
  }
  __syncthreads();

  const int mrow = lane & 15;
  const int hi8  = (lane & 16) ? 8 : 0;

  // ---- Phase 1: qh = q @ Wq  (waves split the 8 N-tiles) ----
  for (int t = 0; t < 2; ++t) {
    const int nt = wave * 2 + t;
    v8f acc = {};
#pragma unroll
    for (int ks = 0; ks < 4; ++ks)
      acc = wmma_f16(load_frag_a(s->qb + ks * 32, DD, lane),
                     load_frag_bT(s->wqT + (nt * 16) * DD + ks * 32, DD, lane),
                     acc);
#pragma unroll
    for (int r = 0; r < 8; ++r)
      s->qhf[(r + hi8) * DD + nt * 16 + mrow] = acc[r];
  }
  __syncthreads();

  WaveBuf& W = s->wb[wave];

  // ---- Phase 2: per-n pipeline, 4 positions per wave ----
  for (int it = 0; it < TILE_N / NWAVES; ++it) {
    const int nl = wave * (TILE_N / NWAVES) + it;
    const int n  = n0 + nl;
    const size_t base = ((size_t)b * NB + n) * KK;

    // (a) stage k tile (float4 load -> packed f16 b64 store), mask, pos MLP-1
    {
      const float4* kp4 = reinterpret_cast<const float4*>(k + base * DD);
      for (int i = lane; i < KK * DD / 4; i += 32) {
        const float4 f = kp4[i];
        v4h h = {(_Float16)f.x, (_Float16)f.y, (_Float16)f.z, (_Float16)f.w};
        *reinterpret_cast<v4h*>(W.kb + i * 4) = h;
      }
      if (lane < KK) W.mk[lane] = mask[base + lane];
      const float* pp = pos + base * 4;
#pragma unroll
      for (int e = 0; e < 8; ++e) {
        const int o = lane + 32 * e;  // 256 = 16 rows x 16 hidden
        const int m = o >> 4, h = o & 15;
        float a = s->pb1[h];
#pragma unroll
        for (int j = 0; j < 4; ++j) a += pp[m * 4 + j] * s->p1[j * HH + h];
        W.gb[m * 32 + h] = (_Float16)fmaxf(a, 0.f);
      }
    }
    lds_wave_fence();

    // per-iteration register state
    int mkr[8];
#pragma unroll
    for (int r = 0; r < 8; ++r) mkr[r] = W.mk[r + hi8];
    float qrow[8];
#pragma unroll
    for (int nt = 0; nt < 8; ++nt) qrow[nt] = s->qhf[nl * DD + nt * 16 + mrow];

    float ph[64];  // ph fragment elements, same (m,col) mapping everywhere
    float vp[64];  // (v + ph) fragment elements

    // (b+c) ph = gb @ P2 + pb2 ; kh = kb @ Wk ; s = kh - qh[nl] + ph
    {
      const v16h af_g = load_frag_a(W.gb, 32, lane);
#pragma unroll
      for (int nt = 0; nt < 8; ++nt) {
        v8f pacc = {};
        pacc = wmma_f16(af_g, load_frag_bT(s->p2T + (nt * 16) * 32, 32, lane),
                        pacc);
        v8f kacc = {};
#pragma unroll
        for (int ks = 0; ks < 4; ++ks)
          kacc = wmma_f16(
              load_frag_a(W.kb + ks * 32, DD, lane),
              load_frag_bT(s->wkT + (nt * 16) * DD + ks * 32, DD, lane), kacc);
        const int col = nt * 16 + mrow;
        const float pb2c = s->pb2[col];
#pragma unroll
        for (int r = 0; r < 8; ++r) {
          const int m = r + hi8;
          const float phv = pacc[r] + pb2c;
          ph[nt * 8 + r] = phv;
          const float kh = kacc[r];
          W.khb[m * DD + col] = (_Float16)kh;
          W.sb[m * DD + col]  = (_Float16)(kh - qrow[nt] + phv);
        }
      }
    }
    lds_wave_fence();

    // (d) v = khb @ Wv ; vp = v + ph (registers)
#pragma unroll
    for (int nt = 0; nt < 8; ++nt) {
      v8f acc = {};
#pragma unroll
      for (int ks = 0; ks < 4; ++ks)
        acc = wmma_f16(
            load_frag_a(W.khb + ks * 32, DD, lane),
            load_frag_bT(s->wvT + (nt * 16) * DD + ks * 32, DD, lane), acc);
#pragma unroll
      for (int r = 0; r < 8; ++r) vp[nt * 8 + r] = acc[r] + ph[nt * 8 + r];
    }

    // (e) h = relu(sb @ A1 + ab1)   (single 16-wide N-tile)
    {
      v8f acc = {};
#pragma unroll
      for (int ks = 0; ks < 4; ++ks)
        acc = wmma_f16(load_frag_a(W.sb + ks * 32, DD, lane),
                       load_frag_bT(s->a1T + ks * 32, DD, lane), acc);
      const float ab1c = s->ab1[mrow];
#pragma unroll
      for (int r = 0; r < 8; ++r)
        W.hb[(r + hi8) * 32 + mrow] = (_Float16)fmaxf(acc[r] + ab1c, 0.f);
    }
    lds_wave_fence();

    // (f) attn = hb @ A2 + ab2; mask; softmax over kv (M-dim: 8 regs +
    //     lane<->lane^16); x_row = sum_kv (v+ph)*attn
    {
      const v16h af_h = load_frag_a(W.hb, 32, lane);
#pragma unroll
      for (int nt = 0; nt < 8; ++nt) {
        v8f acc = {};
        acc = wmma_f16(af_h, load_frag_bT(s->a2T + (nt * 16) * 32, 32, lane),
                       acc);
        const int col = nt * 16 + mrow;
        const float ab2c = s->ab2[col];
        float vals[8];
        float mx = -3.0e38f;
#pragma unroll
        for (int r = 0; r < 8; ++r) {
          float v2 = acc[r] + ab2c;
          if (mkr[r] == 0) v2 = -1.0e9f;
          vals[r] = v2;
          mx = fmaxf(mx, v2);
        }
        mx = fmaxf(mx, __shfl_xor(mx, 16, 32));
        float sum = 0.f;
#pragma unroll
        for (int r = 0; r < 8; ++r) {
          vals[r] = __expf(vals[r] - mx);
          sum += vals[r];
        }
        sum += __shfl_xor(sum, 16, 32);
        const float inv = 1.f / sum;
        float part = 0.f;
#pragma unroll
        for (int r = 0; r < 8; ++r) part += vp[nt * 8 + r] * (vals[r] * inv);
        part += __shfl_xor(part, 16, 32);
        if (lane < 16) s->xb[nl * DD + col] = (_Float16)part;
      }
    }
  }
  __syncthreads();

  // ---- Phase 3: out = x @ Wo + bo  (waves split the 8 N-tiles) ----
  for (int t = 0; t < 2; ++t) {
    const int nt = wave * 2 + t;
    v8f acc = {};
#pragma unroll
    for (int ks = 0; ks < 4; ++ks)
      acc = wmma_f16(load_frag_a(s->xb + ks * 32, DD, lane),
                     load_frag_bT(s->woT + (nt * 16) * DD + ks * 32, DD, lane),
                     acc);
    const int col = nt * 16 + mrow;
    const float bc = s->bos[col];
#pragma unroll
    for (int r = 0; r < 8; ++r)
      out[((size_t)b * NB + (n0 + r + hi8)) * DD + col] = acc[r] + bc;
  }
}

extern "C" void kernel_launch(void* const* d_in, const int* in_sizes, int n_in,
                              void* d_out, int out_size, void* d_ws,
                              size_t ws_size, hipStream_t stream) {
  (void)in_sizes; (void)n_in; (void)out_size; (void)d_ws; (void)ws_size;
  const float* q    = (const float*)d_in[0];
  const float* k    = (const float*)d_in[1];
  const float* pos  = (const float*)d_in[2];
  const int*   mask = (const int*)d_in[3];
  const float* Wq   = (const float*)d_in[4];
  const float* Wk   = (const float*)d_in[5];
  const float* Wv   = (const float*)d_in[6];
  const float* P1   = (const float*)d_in[7];
  const float* pb1  = (const float*)d_in[8];
  const float* P2   = (const float*)d_in[9];
  const float* pb2  = (const float*)d_in[10];
  const float* A1   = (const float*)d_in[11];
  const float* ab1  = (const float*)d_in[12];
  const float* A2   = (const float*)d_in[13];
  const float* ab2  = (const float*)d_in[14];
  const float* Wo   = (const float*)d_in[15];
  const float* bo   = (const float*)d_in[16];
  float* out = (float*)d_out;

  const dim3 grid(BB * (NB / TILE_N));
  const dim3 block(NWAVES * 32);
  const size_t shmem = sizeof(Smem);
  hipLaunchKernelGGL(cas_attn2d_kernel, grid, block, shmem, stream, q, k, pos,
                     mask, Wq, Wk, Wv, P1, pb1, P2, pb2, A1, ab1, A2, ab2, Wo,
                     bo, out);
}
